// QuantizedBlock_3015067042324
// MI455X (gfx1250) — matrix operand order repaired and checked by reading, not compiled, not run on hardware
//
#include <hip/hip_runtime.h>
#include <hip/hip_bf16.h>

typedef __attribute__((ext_vector_type(16))) __bf16 bf16x16;
typedef __attribute__((ext_vector_type(8)))  __bf16 bf16x8;
typedef __attribute__((ext_vector_type(8)))  float  f32x8;

// problem dimensions (compile-time where it matters)
constexpr int CB   = 32;     // batch
constexpr int CN   = 577;    // tokens
constexpr int CC   = 1024;   // channels
constexpr int CHD  = 4096;   // mlp hidden
constexpr int CBH  = 512;    // batch*heads
constexpr int NPAD = 608;    // padded token count (19*32)
constexpr int CM   = CB * CN;

static __device__ __forceinline__ bf16x16 cat8(bf16x8 lo, bf16x8 hi) {
    return __builtin_shufflevector(lo, hi, 0,1,2,3,4,5,6,7,8,9,10,11,12,13,14,15);
}
static __device__ __forceinline__ f32x8 wmma_bf16(bf16x16 a, bf16x16 b, f32x8 c) {
    return __builtin_amdgcn_wmma_f32_16x16x32_bf16(false, a, false, b, (short)0, c, false, false);
}
static __device__ __forceinline__ f32x8 fzero8() {
    f32x8 v;
#pragma unroll
    for (int i = 0; i < 8; ++i) v[i] = 0.0f;
    return v;
}

// ds_swizzle xor-mode lane reduction step (group-of-32: offset[14:10]=xor mask)
template <int MASK>
static __device__ __forceinline__ float swz_xor(float v) {
    return __int_as_float(
        __builtin_amdgcn_ds_swizzle(__float_as_int(v), (MASK << 10) | 0x1F));
}

// CDNA5 async global->LDS copy, 16 bytes per lane, tracked by ASYNCcnt
static __device__ __forceinline__ void async_copy16(unsigned lds_off, const void* gaddr) {
    asm volatile("global_load_async_to_lds_b128 %0, %1, off"
                 :: "v"(lds_off), "v"(gaddr) : "memory");
}
template <int N>
static __device__ __forceinline__ void wait_async() {
    asm volatile("s_wait_asynccnt %0" :: "i"(N) : "memory");
}

// ---------------------------------------------------------------------------
// fp32 -> bf16 flat convert (weights)
// ---------------------------------------------------------------------------
__global__ __launch_bounds__(256)
void f32_to_bf16_kernel(const float* __restrict__ in, __bf16* __restrict__ out, int n) {
    int i = blockIdx.x * 256 + threadIdx.x;
    if (i < n) out[i] = (__bf16)in[i];
}

// ---------------------------------------------------------------------------
// LayerNorm over C=1024, one row per block (256 thr x 4 elems), bf16 out
// ---------------------------------------------------------------------------
__global__ __launch_bounds__(256)
void layernorm_bf16_kernel(const float* __restrict__ x, const float* __restrict__ w,
                           const float* __restrict__ b, __bf16* __restrict__ out) {
    __shared__ float red[256];
    const int row = blockIdx.x;
    const int tid = threadIdx.x;
    const float* xr = x + (size_t)row * CC;
    float4 v = *(const float4*)(xr + tid * 4);

    red[tid] = v.x + v.y + v.z + v.w;
    __syncthreads();
#pragma unroll
    for (int st = 128; st > 0; st >>= 1) {
        if (tid < st) red[tid] += red[tid + st];
        __syncthreads();
    }
    float mean = red[0] * (1.0f / CC);
    __syncthreads();
    float dx = v.x - mean, dy = v.y - mean, dz = v.z - mean, dw = v.w - mean;
    red[tid] = dx * dx + dy * dy + dz * dz + dw * dw;
    __syncthreads();
#pragma unroll
    for (int st = 128; st > 0; st >>= 1) {
        if (tid < st) red[tid] += red[tid + st];
        __syncthreads();
    }
    float rstd = rsqrtf(red[0] * (1.0f / CC) + 1e-6f);
    int c = tid * 4;
    __bf16* orow = out + (size_t)row * CC + c;
    orow[0] = (__bf16)(dx * rstd * w[c + 0] + b[c + 0]);
    orow[1] = (__bf16)(dy * rstd * w[c + 1] + b[c + 1]);
    orow[2] = (__bf16)(dz * rstd * w[c + 2] + b[c + 2]);
    orow[3] = (__bf16)(dw * rstd * w[c + 3] + b[c + 3]);
}

// ---------------------------------------------------------------------------
// WMMA GEMM: Out[M,N] = A[M,K](bf16) @ W[N,K](bf16)^T + bias
// Block tile 128x128, 8 waves (4x2) each computing 32x64, K-step 32.
// Double-buffered LDS staged with async global->LDS (ASYNCcnt).
// EPI: 0 = fp32 out, 1 = fp32 out + residual, 2 = GELU -> bf16 out
// ---------------------------------------------------------------------------
template <int EPI, bool OUT_BF16>
__global__ __launch_bounds__(256)
void gemm_wmma_kernel(const __bf16* __restrict__ A, const __bf16* __restrict__ W,
                      const float* __restrict__ bias, const float* __restrict__ resid,
                      void* __restrict__ outp, int Mdim, int Ndim, int Kdim) {
    constexpr int LS = 40;  // LDS row stride in halves (32 data + 8 pad)
    __shared__ __bf16 As[2][128 * LS];
    __shared__ __bf16 Bs[2][128 * LS];

    const int tid  = threadIdx.x;
    const int lane = tid & 31;
    const int wave = tid >> 5;
    const int wm   = wave & 3;   // 0..3 -> M sub-tile
    const int wn   = wave >> 2;  // 0..1 -> N sub-tile
    const int hf   = lane >> 4;  // half-wave
    const int l15  = lane & 15;
    const int m0 = blockIdx.y * 128;
    const int n0 = blockIdx.x * 128;

    f32x8 acc[2][4];
#pragma unroll
    for (int mi = 0; mi < 2; ++mi)
#pragma unroll
        for (int ni = 0; ni < 4; ++ni) acc[mi][ni] = fzero8();

    // issue async stage of one 128x32 A tile + 128x32 W tile (4 x 16B per thread)
    auto stage = [&](int buf, int kt) {
#pragma unroll
        for (int p = 0; p < 2; ++p) {
            int linear = (p * 256 + tid) * 8;  // 8 halves per thread per pass
            int row = linear >> 5;
            int col = linear & 31;
            int gm = m0 + row;
            if (gm >= Mdim) gm = Mdim - 1;  // clamp: OOB rows never stored
            async_copy16((unsigned)(size_t)&As[buf][row * LS + col],
                         A + (size_t)gm * Kdim + kt + col);
            async_copy16((unsigned)(size_t)&Bs[buf][row * LS + col],
                         W + (size_t)(n0 + row) * Kdim + kt + col);
        }
    };

    const int nkt = Kdim >> 5;
    stage(0, 0);
    for (int it = 0; it < nkt; ++it) {
        const int cur = it & 1;
        if (it + 1 < nkt) {
            stage(cur ^ 1, (it + 1) << 5);
            wait_async<4>();   // the 4 newest (next tile) may remain outstanding
        } else {
            wait_async<0>();
        }
        __syncthreads();       // tile `cur` visible to all waves

        bf16x16 af[2], bf[4];
#pragma unroll
        for (int mi = 0; mi < 2; ++mi) {
            const __bf16* base = &As[cur][(wm * 32 + mi * 16 + l15) * LS];
            af[mi] = cat8(*(const bf16x8*)(base + hf * 16),
                          *(const bf16x8*)(base + hf * 16 + 8));
        }
#pragma unroll
        for (int ni = 0; ni < 4; ++ni) {
            const __bf16* base = &Bs[cur][(wn * 64 + ni * 16 + l15) * LS + hf * 16];
            bf[ni] = cat8(*(const bf16x8*)(base), *(const bf16x8*)(base + 8));
        }
#pragma unroll
        for (int mi = 0; mi < 2; ++mi)
#pragma unroll
            for (int ni = 0; ni < 4; ++ni)
                acc[mi][ni] = wmma_bf16(af[mi], bf[ni], acc[mi][ni]);
        __syncthreads();       // all waves done reading `cur` before it is re-staged
    }

    // epilogue: D layout row = r + 8*hf, col = l15
#pragma unroll
    for (int mi = 0; mi < 2; ++mi)
#pragma unroll
        for (int ni = 0; ni < 4; ++ni) {
            int n = n0 + wn * 64 + ni * 16 + l15;
            float bn = bias[n];
#pragma unroll
            for (int r = 0; r < 8; ++r) {
                int m = m0 + wm * 32 + mi * 16 + r + hf * 8;
                if (m < Mdim) {
                    float v = acc[mi][ni][r] + bn;
                    if (EPI == 1) v += resid[(size_t)m * Ndim + n];
                    if (EPI == 2) v = 0.5f * v * (1.0f + erff(v * 0.70710678118654752f));
                    if (OUT_BF16) ((__bf16*)outp)[(size_t)m * Ndim + n] = (__bf16)v;
                    else          ((float*)outp)[(size_t)m * Ndim + n] = v;
                }
            }
        }
}

// ---------------------------------------------------------------------------
// Pack QKV fp32 [B*N, 3072] -> q[BH,N,64]*scale, k[BH,NPAD,64], vt[BH,64,NPAD]
// ---------------------------------------------------------------------------
__global__ __launch_bounds__(256)
void pack_qkv_kernel(const float* __restrict__ qkv, __bf16* __restrict__ q,
                     __bf16* __restrict__ k, __bf16* __restrict__ vt) {
    int idx = blockIdx.x * 256 + threadIdx.x;
    int dh = idx & 63;
    int n  = (idx >> 6) % NPAD;
    int bh = idx / (NPAD * 64);
    if (bh >= CBH) return;
    int b = bh >> 4, h = bh & 15;
    float kv = 0.0f, vv = 0.0f;
    if (n < CN) {
        const float* base = qkv + ((size_t)(b * CN + n)) * 3072 + h * 64 + dh;
        float qv = base[0] * 0.125f;  // Dh=64 -> scale 1/8
        kv = base[1024];
        vv = base[2048];
        q[((size_t)bh * CN + n) * 64 + dh] = (__bf16)qv;
    }
    k[((size_t)bh * NPAD + n) * 64 + dh] = (__bf16)kv;
    vt[(((size_t)bh) * 64 + dh) * NPAD + n] = (__bf16)vv;
}

// ---------------------------------------------------------------------------
// Flash attention: 1 wave per (b,h, 16-query tile); 32 keys per iteration.
// S = Q·K^T (4 wmma), online softmax (max via ds_swizzle, sum via P·ones wmma),
// P·V (4 wmma). o bf16 [B,N,C].
// ---------------------------------------------------------------------------
__global__ __launch_bounds__(32)
void flash_attn_kernel(const __bf16* __restrict__ q, const __bf16* __restrict__ k,
                       const __bf16* __restrict__ vt, __bf16* __restrict__ o) {
    const int lane = threadIdx.x;
    const int qt = blockIdx.x;   // query tile
    const int bh = blockIdx.y;   // b*16 + h
    const int b = bh >> 4, h = bh & 15;
    const int hf = lane >> 4;
    const int l15 = lane & 15;

    __shared__ __bf16 Pl[16 * 40];

    // all-ones B fragment: row sums of P come from one extra WMMA
    bf16x16 ones;
#pragma unroll
    for (int i = 0; i < 16; ++i) ones[i] = (__bf16)1.0f;

    // Q fragments (A layout), Dh=64 -> 2 fragments of K=32
    int qm = qt * 16 + l15;
    if (qm >= CN) qm = CN - 1;
    const __bf16* qrow = q + ((size_t)bh * CN + qm) * 64;
    bf16x16 aq[2];
#pragma unroll
    for (int c = 0; c < 2; ++c)
        aq[c] = cat8(*(const bf16x8*)(qrow + c * 32 + hf * 16),
                     *(const bf16x8*)(qrow + c * 32 + hf * 16 + 8));

    f32x8 accd[4];
#pragma unroll
    for (int d = 0; d < 4; ++d) accd[d] = fzero8();
    float mrow[8], lrow[8];
#pragma unroll
    for (int r = 0; r < 8; ++r) { mrow[r] = -1e30f; lrow[r] = 0.0f; }

    const int ntiles = (CN + 31) / 32;  // 19
    for (int j = 0; j < ntiles; ++j) {
        const int kb = j * 32;
        // ---- S = Q K^T : two 16x16 tiles over 32 keys ----
        f32x8 s[2];
#pragma unroll
        for (int t = 0; t < 2; ++t) {
            s[t] = fzero8();
            const __bf16* krow = k + ((size_t)bh * NPAD + kb + t * 16 + l15) * 64;
#pragma unroll
            for (int c = 0; c < 2; ++c) {
                bf16x16 bk = cat8(*(const bf16x8*)(krow + c * 32 + hf * 16),
                                  *(const bf16x8*)(krow + c * 32 + hf * 16 + 8));
                s[t] = wmma_bf16(aq[c], bk, s[t]);
            }
            int col = kb + t * 16 + l15;  // mask out-of-range keys
            if (col >= CN) {
#pragma unroll
                for (int r = 0; r < 8; ++r) s[t][r] = -1e30f;
            }
        }
        // ---- running max (row = r + 8*hf; xor-swizzle across 16-lane group) ----
        float alpha[8];
#pragma unroll
        for (int r = 0; r < 8; ++r) {
            float v = fmaxf(s[0][r], s[1][r]);
            v = fmaxf(v, swz_xor<8>(v));
            v = fmaxf(v, swz_xor<4>(v));
            v = fmaxf(v, swz_xor<2>(v));
            v = fmaxf(v, swz_xor<1>(v));
            float mnew = fmaxf(mrow[r], v);
            alpha[r] = __expf(mrow[r] - mnew);
            mrow[r] = mnew;
            float p0 = __expf(s[0][r] - mnew);
            float p1 = __expf(s[1][r] - mnew);
            int row = r + hf * 8;  // bounce P to LDS (D layout -> A layout)
            Pl[row * 40 + l15]      = (__bf16)p0;
            Pl[row * 40 + 16 + l15] = (__bf16)p1;
        }
        __syncthreads();
        const __bf16* pr = &Pl[l15 * 40];
        bf16x16 ap = cat8(*(const bf16x8*)(pr + hf * 16),
                          *(const bf16x8*)(pr + hf * 16 + 8));
        __syncthreads();
        // ---- row sums via P x ones (replicated across lanes in D layout) ----
        f32x8 rsum = wmma_bf16(ap, ones, fzero8());
#pragma unroll
        for (int r = 0; r < 8; ++r) {
            lrow[r] = lrow[r] * alpha[r] + rsum[r];
#pragma unroll
            for (int d = 0; d < 4; ++d) accd[d][r] *= alpha[r];
        }
        // ---- O += P V : 4 wmma over Dh chunks; vt contiguous along keys ----
#pragma unroll
        for (int d = 0; d < 4; ++d) {
            const __bf16* vrow = vt + (((size_t)bh) * 64 + d * 16 + l15) * NPAD + kb + hf * 16;
            bf16x16 bv = cat8(*(const bf16x8*)(vrow), *(const bf16x8*)(vrow + 8));
            accd[d] = wmma_bf16(ap, bv, accd[d]);
        }
    }
    // ---- normalize + store o[b, m, h*64 + dh] ----
#pragma unroll
    for (int d = 0; d < 4; ++d)
#pragma unroll
        for (int r = 0; r < 8; ++r) {
            int m = qt * 16 + r + hf * 8;
            if (m < CN) {
                float v = accd[d][r] / lrow[r];
                o[((size_t)(b * CN + m)) * CC + h * 64 + d * 16 + l15] = (__bf16)v;
            }
        }
}

// ---------------------------------------------------------------------------
extern "C" void kernel_launch(void* const* d_in, const int* in_sizes, int n_in,
                              void* d_out, int out_size, void* d_ws, size_t ws_size,
                              hipStream_t stream) {
    const int C = CC, Hd = CHD, M = CM;

    const float* x      = (const float*)d_in[0];
    const float* ln1_w  = (const float*)d_in[1];
    const float* ln1_b  = (const float*)d_in[2];
    const float* qkv_w  = (const float*)d_in[3];
    const float* qkv_b  = (const float*)d_in[4];
    const float* proj_w = (const float*)d_in[5];
    const float* proj_b = (const float*)d_in[6];
    const float* ln2_w  = (const float*)d_in[7];
    const float* ln2_b  = (const float*)d_in[8];
    const float* fc1_w  = (const float*)d_in[9];
    const float* fc1_b  = (const float*)d_in[10];
    const float* fc2_w  = (const float*)d_in[11];
    const float* fc2_b  = (const float*)d_in[12];
    float* out = (float*)d_out;

    // workspace sub-allocation (256B aligned)
    char* ws = (char*)d_ws;
    size_t off = 0;
    auto alloc = [&](size_t bytes) { size_t o = off; off += (bytes + 255) & ~(size_t)255; return o; };
    size_t o_wqkv  = alloc((size_t)3 * C * C * 2);
    size_t o_wproj = alloc((size_t)C * C * 2);
    size_t o_wfc1  = alloc((size_t)Hd * C * 2);
    size_t o_wfc2  = alloc((size_t)C * Hd * 2);
    size_t o_act   = alloc((size_t)M * C * 2);        // ln1 out, later attention out
    size_t o_big   = alloc((size_t)M * 3 * C * 4);    // qkv fp32, later ln2 + gelu bf16
    size_t o_q     = alloc((size_t)CBH * CN * 64 * 2);
    size_t o_k     = alloc((size_t)CBH * NPAD * 64 * 2);
    size_t o_vt    = alloc((size_t)CBH * NPAD * 64 * 2);
    size_t o_x1    = alloc((size_t)M * C * 4);

    __bf16* wqkv  = (__bf16*)(ws + o_wqkv);
    __bf16* wproj = (__bf16*)(ws + o_wproj);
    __bf16* wfc1  = (__bf16*)(ws + o_wfc1);
    __bf16* wfc2  = (__bf16*)(ws + o_wfc2);
    __bf16* actb  = (__bf16*)(ws + o_act);
    float*  qkvf  = (float*)(ws + o_big);
    __bf16* ln2b  = (__bf16*)(ws + o_big);                         // reuse after pack
    __bf16* gelub = (__bf16*)(ws + o_big + (size_t)M * C * 2);     // fits inside qkv region
    __bf16* qb    = (__bf16*)(ws + o_q);
    __bf16* kbuf  = (__bf16*)(ws + o_k);
    __bf16* vtb   = (__bf16*)(ws + o_vt);
    float*  x1    = (float*)(ws + o_x1);

    auto cvt = [&](const float* src, __bf16* dst, int n) {
        f32_to_bf16_kernel<<<(n + 255) / 256, 256, 0, stream>>>(src, dst, n);
    };
    cvt(qkv_w, wqkv, 3 * C * C);
    cvt(proj_w, wproj, C * C);
    cvt(fc1_w, wfc1, Hd * C);
    cvt(fc2_w, wfc2, C * Hd);

    const int gm = (M + 127) / 128;  // 145

    // LN1 -> bf16
    layernorm_bf16_kernel<<<M, 256, 0, stream>>>(x, ln1_w, ln1_b, actb);

    // QKV GEMM: [M,1024] x [3072,1024]^T -> fp32
    gemm_wmma_kernel<0, false><<<dim3(3 * C / 128, gm), 256, 0, stream>>>(
        actb, wqkv, qkv_b, nullptr, qkvf, M, 3 * C, C);

    // pack q/k/vt
    pack_qkv_kernel<<<(CBH * NPAD * 64 + 255) / 256, 256, 0, stream>>>(qkvf, qb, kbuf, vtb);

    // flash attention -> actb (bf16 [M, C])
    flash_attn_kernel<<<dim3((CN + 15) / 16, CBH), 32, 0, stream>>>(qb, kbuf, vtb, actb);

    // proj GEMM + residual(x) -> x1 fp32
    gemm_wmma_kernel<1, false><<<dim3(C / 128, gm), 256, 0, stream>>>(
        actb, wproj, proj_b, x, x1, M, C, C);

    // LN2 -> bf16
    layernorm_bf16_kernel<<<M, 256, 0, stream>>>(x1, ln2_w, ln2_b, ln2b);

    // FC1 GEMM + GELU -> bf16
    gemm_wmma_kernel<2, true><<<dim3(Hd / 128, gm), 256, 0, stream>>>(
        ln2b, wfc1, fc1_b, nullptr, gelub, M, Hd, C);

    // FC2 GEMM + residual(x1) -> d_out fp32
    gemm_wmma_kernel<1, false><<<dim3(C / 128, gm), 256, 0, stream>>>(
        gelub, wfc2, fc2_b, x1, out, M, C, Hd);
}
